// APPNP_6828998001546
// MI455X (gfx1250) — matrix operand order repaired and checked by reading, not compiled, not run on hardware
//
#include <hip/hip_runtime.h>

typedef __attribute__((ext_vector_type(16))) _Float16 v16h;
typedef __attribute__((ext_vector_type(8)))  _Float16 v8h;
typedef __attribute__((ext_vector_type(8)))  float    v8f;
typedef __attribute__((ext_vector_type(4)))  float    v4f;

#define N_NODES 100000
#define N_EDGES 3200000
#define IN_CH   500
#define HID_CH  64
#define OUT_CH  16
#define KPAD    512
#define W1_STRIDE 520   // halves per column: 512 + 8 pad  -> conflict-free LDS banks
#define W2_STRIDE 72    // 64 + 8 pad
#define H_STRIDE  72

// ---------------------------------------------------------------------------
// Fused MLP: h = relu(x@W1 + b1) @ W2 + b2, via v_wmma_f32_16x16x32_f16.
// One wave computes a 16-row tile; 8 waves / block; grid-stride over tiles.
// ---------------------------------------------------------------------------
__global__ __launch_bounds__(256) void mlp_wmma_kernel(
    const float* __restrict__ x,  const float* __restrict__ W1,
    const float* __restrict__ b1, const float* __restrict__ W2,
    const float* __restrict__ b2, float* __restrict__ h,
    float* __restrict__ h0)
{
  __shared__ __attribute__((aligned(32))) _Float16 w1s[HID_CH * W1_STRIDE];
  __shared__ __attribute__((aligned(32))) _Float16 w2s[OUT_CH * W2_STRIDE];
  __shared__ __attribute__((aligned(32))) _Float16 hs[8 * 16 * H_STRIDE];

  const int tid = threadIdx.x;

  // Stage W1 -> LDS f16, column-major (k contiguous), zero-pad K to 512.
  for (int idx = tid; idx < KPAD * HID_CH; idx += 256) {
    int k = idx >> 6;          // / 64
    int n = idx & 63;
    float v = (k < IN_CH) ? W1[k * HID_CH + n] : 0.0f;
    w1s[n * W1_STRIDE + k] = (_Float16)v;
  }
  // Stage W2 -> LDS f16, column-major.
  for (int idx = tid; idx < HID_CH * OUT_CH; idx += 256) {
    int k = idx >> 4;          // / 16
    int n = idx & 15;
    w2s[n * W2_STRIDE + k] = (_Float16)W2[k * OUT_CH + n];
  }
  __syncthreads();

  const int wave  = tid >> 5;
  const int lane  = tid & 31;
  const int l16   = lane & 15;
  const bool hiL  = (lane >= 16);
  const int hiK8  = hiL ? 8 : 0;
  const int hiK16 = hiL ? 16 : 0;

  const float b2c = b2[l16];
  float b1c[4];
  #pragma unroll
  for (int nt = 0; nt < 4; ++nt) b1c[nt] = b1[nt * 16 + l16];

  const int nTiles = N_NODES / 16;   // 6250 (exact)
  for (int tileBase = blockIdx.x * 8; tileBase < nTiles; tileBase += gridDim.x * 8) {
    const int tile = tileBase + wave;
    if (tile < nTiles) {
      const int rowBase = tile * 16;
      const int row     = rowBase + l16;
      const float* xr   = x + (size_t)row * IN_CH;

      v8f acc[4] = {};

      for (int ks = 0; ks < 16; ++ks) {
        // --- A fragment (16x32 f16): convert x fp32 -> f16 in registers ---
        v16h a;
        const int kb = ks * 32 + hiK8;
        if (ks < 15) {                       // fully in-range, vector loads
          v4f f0 = *(const v4f*)(xr + kb);
          v4f f1 = *(const v4f*)(xr + kb + 4);
          v4f f2 = *(const v4f*)(xr + kb + 16);
          v4f f3 = *(const v4f*)(xr + kb + 20);
          #pragma unroll
          for (int i = 0; i < 4; ++i) {
            a[i]      = (_Float16)f0[i];
            a[4 + i]  = (_Float16)f1[i];
            a[8 + i]  = (_Float16)f2[i];
            a[12 + i] = (_Float16)f3[i];
          }
        } else {                             // K tail 480..511: mask past 500
          #pragma unroll
          for (int i = 0; i < 8; ++i) {
            int k0 = kb + i;
            int k1 = kb + 16 + i;
            a[i]     = (_Float16)((k0 < IN_CH) ? xr[k0] : 0.0f);
            a[8 + i] = (_Float16)((k1 < IN_CH) ? xr[k1] : 0.0f);
          }
        }
        // --- 4 column tiles of W1 (B fragments from LDS) ---
        #pragma unroll
        for (int nt = 0; nt < 4; ++nt) {
          const _Float16* bp = &w1s[(nt * 16 + l16) * W1_STRIDE + ks * 32 + hiK8];
          v8h blo = *(const v8h*)bp;
          v8h bhi = *(const v8h*)(bp + 16);
          v16h b;
          #pragma unroll
          for (int i = 0; i < 8; ++i) { b[i] = blo[i]; b[8 + i] = bhi[i]; }
          acc[nt] = __builtin_amdgcn_wmma_f32_16x16x32_f16(
              false, a, false, b, (short)0, acc[nt], false, false);
        }
      }

      // bias + relu, stage hidden tile (16x64 f16) to per-wave LDS
      _Float16* hw = &hs[wave * 16 * H_STRIDE];
      #pragma unroll
      for (int nt = 0; nt < 4; ++nt) {
        #pragma unroll
        for (int j = 0; j < 8; ++j) {
          float v = acc[nt][j] + b1c[nt];
          v = v > 0.0f ? v : 0.0f;
          const int m = j + (hiL ? 8 : 0);
          hw[m * H_STRIDE + nt * 16 + l16] = (_Float16)v;
        }
      }
      // DS ops from one wave complete in order -> same-wave cross-lane RAW ok.

      // --- GEMM2: hidden(16x64) @ W2(64x16), 2 WMMA k-steps ---
      v8f c2 = {};
      #pragma unroll
      for (int ks = 0; ks < 2; ++ks) {
        const _Float16* ap = &hw[l16 * H_STRIDE + ks * 32];
        v8h a0 = *(const v8h*)(ap + hiK8);
        v8h a1 = *(const v8h*)(ap + 16 + hiK8);
        v16h a;
        #pragma unroll
        for (int i = 0; i < 8; ++i) { a[i] = a0[i]; a[8 + i] = a1[i]; }
        const _Float16* bp = &w2s[l16 * W2_STRIDE + ks * 32 + hiK8];
        v8h blo = *(const v8h*)bp;
        v8h bhi = *(const v8h*)(bp + 16);
        v16h b;
        #pragma unroll
        for (int i = 0; i < 8; ++i) { b[i] = blo[i]; b[8 + i] = bhi[i]; }
        c2 = __builtin_amdgcn_wmma_f32_16x16x32_f16(
            false, a, false, b, (short)0, c2, false, false);
      }
      #pragma unroll
      for (int j = 0; j < 8; ++j) {
        const int m = rowBase + j + (hiL ? 8 : 0);
        const float v = c2[j] + b2c;
        h [(size_t)m * OUT_CH + l16] = v;
        h0[(size_t)m * OUT_CH + l16] = v;
      }
    }
  }
}

// ---------------------------------------------------------------------------
// GCN normalization + APPNP propagation (L2-resident scatter/gather)
// ---------------------------------------------------------------------------
__global__ void deg_init_kernel(float* __restrict__ deg) {
  int i = blockIdx.x * 256 + threadIdx.x;
  if (i < N_NODES) deg[i] = 1.0f;            // self-loop
}

__global__ void deg_count_kernel(const long long* __restrict__ col,
                                 float* __restrict__ deg) {
  int e = blockIdx.x * 256 + threadIdx.x;
  if (e < N_EDGES) atomicAdd(&deg[(int)col[e]], 1.0f);
}

__global__ void dinv_kernel(const float* __restrict__ deg,
                            float* __restrict__ dinv) {
  int i = blockIdx.x * 256 + threadIdx.x;
  if (i < N_NODES) dinv[i] = rsqrtf(deg[i]); // deg >= 1 always
}

// tmp = self-loop contribution: dinv[i]^2 * h[i]
__global__ void self_init_kernel(const float* __restrict__ h,
                                 const float* __restrict__ dinv,
                                 float* __restrict__ tmp) {
  int idx = blockIdx.x * 256 + threadIdx.x;
  if (idx < N_NODES * OUT_CH) {
    int i = idx >> 4;
    float d = dinv[i];
    tmp[idx] = d * d * h[idx];
  }
}

// tmp[col] += dinv[row]*dinv[col] * h[row]   (one thread per edge-channel)
__global__ void scatter_kernel(const long long* __restrict__ row,
                               const long long* __restrict__ col,
                               const float* __restrict__ dinv,
                               const float* __restrict__ h,
                               float* __restrict__ tmp) {
  int t = blockIdx.x * 256 + threadIdx.x;
  if (t < N_EDGES * OUT_CH) {
    int e = t >> 4;
    int c = t & 15;
    int r  = (int)row[e];
    int cl = (int)col[e];
    float w = dinv[r] * dinv[cl];
    atomicAdd(&tmp[cl * OUT_CH + c], w * h[r * OUT_CH + c]);
  }
}

// h = (1-alpha)*tmp + alpha*h0
__global__ void final_kernel(const float* __restrict__ tmp,
                             const float* __restrict__ h0,
                             float* __restrict__ h) {
  int idx = blockIdx.x * 256 + threadIdx.x;
  if (idx < N_NODES * OUT_CH) h[idx] = 0.9f * tmp[idx] + 0.1f * h0[idx];
}

// ---------------------------------------------------------------------------
extern "C" void kernel_launch(void* const* d_in, const int* in_sizes, int n_in,
                              void* d_out, int out_size, void* d_ws, size_t ws_size,
                              hipStream_t stream) {
  const float*     x  = (const float*)d_in[0];
  const long long* ei = (const long long*)d_in[1];   // int64 edge_index [2, E]
  const float*     W1 = (const float*)d_in[2];
  const float*     b1 = (const float*)d_in[3];
  const float*     W2 = (const float*)d_in[4];
  const float*     b2 = (const float*)d_in[5];

  float* h = (float*)d_out;                  // [N,16], propagated in-place

  float* h0   = (float*)d_ws;                // N*16
  float* tmp  = h0  + (size_t)N_NODES * OUT_CH;
  float* deg  = tmp + (size_t)N_NODES * OUT_CH;
  float* dinv = deg + N_NODES;

  const long long* rowIdx = ei;              // sources
  const long long* colIdx = ei + N_EDGES;    // targets

  // MLP (writes h and h0)
  mlp_wmma_kernel<<<640, 256, 0, stream>>>(x, W1, b1, W2, b2, h, h0);

  // degrees / normalization
  const int nb  = (N_NODES + 255) / 256;
  const int eb  = (N_EDGES + 255) / 256;
  const int nhb = (N_NODES * OUT_CH + 255) / 256;
  const int ehb = (N_EDGES * OUT_CH + 255) / 256;
  deg_init_kernel <<<nb, 256, 0, stream>>>(deg);
  deg_count_kernel<<<eb, 256, 0, stream>>>(colIdx, deg);
  dinv_kernel     <<<nb, 256, 0, stream>>>(deg, dinv);

  // K = 10 APPNP steps
  for (int k = 0; k < 10; ++k) {
    self_init_kernel<<<nhb, 256, 0, stream>>>(h, dinv, tmp);
    scatter_kernel  <<<ehb, 256, 0, stream>>>(rowIdx, colIdx, dinv, h, tmp);
    final_kernel    <<<nhb, 256, 0, stream>>>(tmp, h0, h);
  }
}
